// MultiHeadAttention_22926535426625
// MI455X (gfx1250) — matrix operand order repaired and checked by reading, not compiled, not run on hardware
//
#include <hip/hip_runtime.h>
#include <hip/hip_bf16.h>

// ---------------------------------------------------------------------------
// Transformer attention block (B=4,S=2048,D=1024,H=16,DH=64) for gfx1250.
// GEMMs + flash attention on v_wmma_f32_16x16x32_bf16 (wave32).
// ALL mainloop global traffic goes through GLOBAL_LOAD_ASYNC_TO_LDS_B128
// (ASYNCcnt, double-buffered); WMMAs are fed exclusively by ds_load_b128,
// so the only loop waits are explicit s_wait_asynccnt and partial dscnt.
// ---------------------------------------------------------------------------

#define Bsz 4
#define Ssz 2048
#define Dsz 1024
#define Hsz 16
#define DHsz 64
#define Msz (Bsz * Ssz)          // 8192 rows
#define LN_EPS 1e-5f

typedef __attribute__((ext_vector_type(16))) __bf16 v16bf;
typedef __attribute__((ext_vector_type(8)))  __bf16 v8bf;
typedef __attribute__((ext_vector_type(8)))  float  v8f;

static __device__ inline v16bf combine16(v8bf lo, v8bf hi) {
    return __builtin_shufflevector(lo, hi, 0,1,2,3,4,5,6,7,8,9,10,11,12,13,14,15);
}

// A-fragment (16x32 bf16, MxK): lane m = lane&15, half h = lane>>4.
// halves j=0..7 -> K = h*8 + j ; j=8..15 -> K = 16 + h*8 + (j-8)
static __device__ inline v16bf load_a_frag(const __bf16* rowbase, int h) {
    v8bf lo = *(const v8bf*)(rowbase + h * 8);
    v8bf hi = *(const v8bf*)(rowbase + 16 + h * 8);
    return combine16(lo, hi);
}

// B-fragment (32x16 bf16, KxN): lane n = lane&15, half h = lane>>4,
// halves j=0..7 -> K = h*8 + j ; j=8..15 -> K = 16 + h*8 + (j-8) (two 16B ds_load_b128)
static __device__ inline v16bf load_b_frag_lds(const __bf16* rowbase, int h) {
    v8bf lo = *(const v8bf*)(rowbase + h * 8);
    v8bf hi = *(const v8bf*)(rowbase + 16 + h * 8);
    return combine16(lo, hi);
}

static __device__ inline v8f wmma_bf16(v16bf a, v16bf b, v8f c) {
    return __builtin_amdgcn_wmma_f32_16x16x32_bf16(false, a, false, b, (short)0, c,
                                                   false, false);
}

// LDS byte-offset of a __shared__ object (flat LDS aperture keeps the offset
// in the low 32 bits).
static __device__ inline unsigned lds_off(const void* p) {
    return (unsigned)(unsigned long long)p;
}

// One 16-byte async DMA transfer global -> LDS (tracked by ASYNCcnt).
static __device__ inline void async_cp16(unsigned ldsDst, const void* gsrc) {
    unsigned long long ga = (unsigned long long)gsrc;
    asm volatile("global_load_async_to_lds_b128 %0, %1, off"
                 :: "v"(ldsDst), "v"(ga) : "memory");
}

#define WAIT_ASYNC(n) asm volatile("s_wait_asynccnt " #n ::: "memory")
#define WAIT_DS0()    asm volatile("s_wait_dscnt 0x0" ::: "memory")

// ---------------------------------------------------------------------------
// f32 -> bf16 conversion (grid-stride)
// ---------------------------------------------------------------------------
__global__ void k_f32_to_bf16(const float* __restrict__ src, __bf16* __restrict__ dst,
                              long n) {
    long i = (long)blockIdx.x * blockDim.x + threadIdx.x;
    long stride = (long)gridDim.x * blockDim.x;
    for (; i < n; i += stride) dst[i] = (__bf16)src[i];
}

// ---------------------------------------------------------------------------
// Shared GEMM mainloop: acc[4] (16M x 64N per wave) += X * W^T over K=1024.
//   abuf: 128M x 32K bf16 slab  (rows padded to 40 halves, double buffered)
//   bbuf:  64N x 32K bf16 slab  (rows padded to 40 halves, double buffered)
// Per k-step each thread issues 3 async 16B DMAs (2 for A, 1 for B); mainloop
// has zero direct global loads.
// ---------------------------------------------------------------------------
static __device__ inline void gemm_mainloop(const __bf16* __restrict__ X,
                                            const __bf16* __restrict__ W,
                                            int mblk, int n0, int wave, int lm, int hh,
                                            __bf16* abuf, __bf16* bbuf, v8f acc[4]) {
    const int tid = threadIdx.x;
    // B copy: 64 rows x 4 chunks of 16B
    const int brow = tid >> 2, bchk = tid & 3;
    const __bf16* bsrc = W + (size_t)(n0 + brow) * Dsz + bchk * 8;
    const unsigned bdst[2] = {
        lds_off(bbuf + (size_t)brow * 40 + bchk * 8),
        lds_off(bbuf + (size_t)(64 + brow) * 40 + bchk * 8)};
    // A copy: 128 rows x 2x(2 chunks of 16B) -> 2 DMAs per thread
    const int arow_c = tid >> 1, ahlf = tid & 1;
    const __bf16* asrc = X + (size_t)(mblk + arow_c) * Dsz + ahlf * 16;
    const unsigned adst[2] = {
        lds_off(abuf + (size_t)arow_c * 40 + ahlf * 16),
        lds_off(abuf + (size_t)(128 + arow_c) * 40 + ahlf * 16)};

    // preload k0 = 0 into buffer 0
    async_cp16(bdst[0], bsrc);
    async_cp16(adst[0], asrc);
    async_cp16(adst[0] + 16, asrc + 8);

    for (int k0 = 0; k0 < Dsz; k0 += 32) {
        const int buf = (k0 >> 5) & 1;
        const int knx = (k0 + 32 < Dsz) ? (k0 + 32) : 0;  // last iter: harmless dummy
        async_cp16(bdst[buf ^ 1], bsrc + knx);
        async_cp16(adst[buf ^ 1], asrc + knx);
        async_cp16(adst[buf ^ 1] + 16, asrc + knx + 8);
        WAIT_ASYNC(0x3);                                  // current buffer landed
        __syncthreads();

        const __bf16* amy = abuf + (size_t)(buf * 128 + wave * 16 + lm) * 40;
        v8bf alo = *(const v8bf*)(amy + hh * 8);
        v8bf ahi = *(const v8bf*)(amy + 16 + hh * 8);
        v16bf a = combine16(alo, ahi);
#pragma unroll
        for (int t = 0; t < 4; ++t) {
            const __bf16* brow2 = bbuf + (size_t)(buf * 64 + t * 16 + lm) * 40;
            acc[t] = wmma_bf16(a, load_b_frag_lds(brow2, hh), acc[t]);
        }
        __syncthreads();                                  // reads done before overwrite
    }
}

// ---------------------------------------------------------------------------
// QKV projection: Y[m,e] = sum_k X[m,k]*W[e,k] + bias[e]
// mode 0: store [B,H,S,DH] (Q,K);  mode 1: store [B,H,DH,S] (V transposed)
// grid = (M/128, D/64), 256 threads (8 waves along M)
// ---------------------------------------------------------------------------
__global__ void k_gemm_qkv(const __bf16* __restrict__ X,
                           const __bf16* __restrict__ W,
                           const float*  __restrict__ bias,
                           __bf16* __restrict__ Y, int mode) {
    __shared__ __bf16 abuf[2 * 128 * 40];
    __shared__ __bf16 bbuf[2 * 64 * 40];
    const int lane = threadIdx.x & 31, wave = threadIdx.x >> 5;
    const int lm = lane & 15, hh = lane >> 4;
    const int mblk = blockIdx.x * 128;
    const int m0 = mblk + wave * 16;
    const int n0 = blockIdx.y * 64;

    v8f acc[4] = {v8f{}, v8f{}, v8f{}, v8f{}};
    gemm_mainloop(X, W, mblk, n0, wave, lm, hh, abuf, bbuf, acc);

#pragma unroll
    for (int t = 0; t < 4; ++t) {
#pragma unroll
        for (int r = 0; r < 8; ++r) {
            int m = m0 + hh * 8 + r;
            int e = n0 + t * 16 + lm;
            int b = m / Ssz, s = m % Ssz;
            int hd = e >> 6, dh = e & 63;
            float v = acc[t][r] + bias[e];
            size_t idx = (mode == 0)
                ? ((((size_t)b * Hsz + hd) * Ssz + s) * DHsz + dh)
                : ((((size_t)b * Hsz + hd) * DHsz + dh) * Ssz + s);
            Y[idx] = (__bf16)v;
        }
    }
}

// ---------------------------------------------------------------------------
// Output projection + bias + residual (f32 out)
// ---------------------------------------------------------------------------
__global__ void k_gemm_oproj(const __bf16* __restrict__ Hc,
                             const __bf16* __restrict__ Wo,
                             const float*  __restrict__ bo,
                             const float*  __restrict__ x,
                             float* __restrict__ resid) {
    __shared__ __bf16 abuf[2 * 128 * 40];
    __shared__ __bf16 bbuf[2 * 64 * 40];
    const int lane = threadIdx.x & 31, wave = threadIdx.x >> 5;
    const int lm = lane & 15, hh = lane >> 4;
    const int mblk = blockIdx.x * 128;
    const int m0 = mblk + wave * 16;
    const int n0 = blockIdx.y * 64;

    v8f acc[4] = {v8f{}, v8f{}, v8f{}, v8f{}};
    gemm_mainloop(Hc, Wo, mblk, n0, wave, lm, hh, abuf, bbuf, acc);

#pragma unroll
    for (int t = 0; t < 4; ++t) {
#pragma unroll
        for (int r = 0; r < 8; ++r) {
            int m = m0 + hh * 8 + r;
            int e = n0 + t * 16 + lm;
            size_t idx = (size_t)m * Dsz + e;
            resid[idx] = x[idx] + acc[t][r] + bo[e];
        }
    }
}

// ---------------------------------------------------------------------------
// Flash attention. grid = B*H*(S/128), 256 threads = 8 waves, one 16-q tile
// each; all 8 waves share one (b,h): K/V tiles staged once per block via
// async DMA (double-buffered); mask row pre-staged in LDS as f32 bias so the
// kv loop has no direct global loads at all.
// ---------------------------------------------------------------------------
__global__ void k_flash_attn(const __bf16* __restrict__ Q,
                             const __bf16* __restrict__ K,
                             const __bf16* __restrict__ Vt,
                             const int*    __restrict__ mask,
                             __bf16* __restrict__ Hctx) {
    __shared__ __bf16 kbuf[2 * 32 * 72];   // 32kv x 64dh, rows padded to 72
    __shared__ __bf16 vbuf[2 * 64 * 40];   // 64dh x 32kv, rows padded to 40
    __shared__ __bf16 pbuf[8 * 16 * 40];   // per-wave P tile bounce
    __shared__ float  mbias[Ssz];          // mask -> additive bias, staged once

    const int tid  = threadIdx.x;
    const int lane = tid & 31, wave = tid >> 5;
    const int lm = lane & 15, hh = lane >> 4;

    const int bh   = blockIdx.x >> 4;
    const int qblk = blockIdx.x & 15;
    const int b    = bh / Hsz;
    const int head = bh % Hsz;
    const int q0   = qblk * 128 + wave * 16;

    const __bf16* Qb  = Q  + (size_t)bh * Ssz * DHsz;
    const __bf16* Kb  = K  + (size_t)bh * Ssz * DHsz;
    const __bf16* Vtb = Vt + (size_t)bh * DHsz * Ssz;
    const int*    mrow_g = mask + (size_t)b * Ssz;

    // stage mask row once
    for (int i = tid; i < Ssz; i += 256)
        mbias[i] = (mrow_g[i] == 0) ? -1.0e30f : 0.f;

    // cooperative-copy mappings (one 16B chunk per thread per tile)
    const int krow = tid >> 3, kchk = tid & 7;   // K: 32 rows x 8 chunks
    const int vrow = tid >> 2, vchk = tid & 3;   // Vt: 64 rows x 4 chunks
    const __bf16* ksrc = Kb + (size_t)krow * DHsz + kchk * 8;
    const __bf16* vsrc = Vtb + (size_t)vrow * Ssz + vchk * 8;
    const unsigned kdst[2] = {lds_off(kbuf + (size_t)krow * 72 + kchk * 8),
                              lds_off(kbuf + (size_t)(32 + krow) * 72 + kchk * 8)};
    const unsigned vdst[2] = {lds_off(vbuf + (size_t)vrow * 40 + vchk * 8),
                              lds_off(vbuf + (size_t)(64 + vrow) * 40 + vchk * 8)};

    // Q fragments resident for the whole KV loop
    const __bf16* qrow = Qb + (size_t)(q0 + lm) * DHsz;
    v16bf qa[2];
    qa[0] = load_a_frag(qrow, hh);
    qa[1] = load_a_frag(qrow + 32, hh);

    v8f oacc[4] = {v8f{}, v8f{}, v8f{}, v8f{}};
    float mrow[8], lrow[8];
#pragma unroll
    for (int r = 0; r < 8; ++r) { mrow[r] = -3.0e38f; lrow[r] = 0.f; }

    const float sc_inv = 0.125f;   // 1/sqrt(64)
    __bf16* mypb = pbuf + (size_t)wave * 16 * 40;

    // preload kv0 = 0 into buffer 0
    async_cp16(kdst[0], ksrc);
    async_cp16(vdst[0], vsrc);
    __syncthreads();               // mbias staged

    for (int kv0 = 0; kv0 < Ssz; kv0 += 32) {
        const int buf = (kv0 >> 5) & 1;
        const int kvn = (kv0 + 32 < Ssz) ? (kv0 + 32) : 0;   // dummy on last iter
        async_cp16(kdst[buf ^ 1], ksrc + (size_t)kvn * DHsz);
        async_cp16(vdst[buf ^ 1], vsrc + kvn);
        WAIT_ASYNC(0x2);                                     // current pair landed
        __syncthreads();

        // ---- scores: 16q x 32kv, contraction over dh=64
        v8f sc[2] = {v8f{}, v8f{}};
#pragma unroll
        for (int kk = 0; kk < 2; ++kk) {
#pragma unroll
            for (int t = 0; t < 2; ++t) {
                const __bf16* brow =
                    kbuf + (size_t)(buf * 32 + t * 16 + lm) * 72 + kk * 32;
                sc[t] = wmma_bf16(qa[kk], load_b_frag_lds(brow, hh), sc[t]);
            }
        }
        float mb0 = mbias[kv0 + lm];
        float mb1 = mbias[kv0 + 16 + lm];

        // ---- online softmax (rows hh*8+r live on this half-wave)
        float p0v[8], p1v[8];
#pragma unroll
        for (int r = 0; r < 8; ++r) {
            float s0 = sc[0][r] * sc_inv + mb0;
            float s1 = sc[1][r] * sc_inv + mb1;
            float mx = fmaxf(s0, s1);
            for (int o = 1; o < 16; o <<= 1) mx = fmaxf(mx, __shfl_xor(mx, o, 32));
            float mnew = fmaxf(mrow[r], mx);
            float corr = __expf(mrow[r] - mnew);
            float p0 = __expf(s0 - mnew);
            float p1 = __expf(s1 - mnew);
            float rs = p0 + p1;
            for (int o = 1; o < 16; o <<= 1) rs += __shfl_xor(rs, o, 32);
            lrow[r] = lrow[r] * corr + rs;
            mrow[r] = mnew;
#pragma unroll
            for (int t2 = 0; t2 < 4; ++t2) oacc[t2][r] *= corr;
            p0v[r] = p0; p1v[r] = p1;
        }

        // ---- P: D-layout -> A-layout via per-wave LDS bounce
#pragma unroll
        for (int r = 0; r < 8; ++r) {
            mypb[(size_t)(hh * 8 + r) * 40 + lm]      = (__bf16)p0v[r];
            mypb[(size_t)(hh * 8 + r) * 40 + 16 + lm] = (__bf16)p1v[r];
        }
        WAIT_DS0();
        v8bf plo = *(const v8bf*)(mypb + (size_t)lm * 40 + hh * 8);
        v8bf phi = *(const v8bf*)(mypb + (size_t)lm * 40 + 16 + hh * 8);
        v16bf pa = combine16(plo, phi);

        // ---- O += P * V
#pragma unroll
        for (int t2 = 0; t2 < 4; ++t2) {
            const __bf16* brow = vbuf + (size_t)(buf * 64 + t2 * 16 + lm) * 40;
            oacc[t2] = wmma_bf16(pa, load_b_frag_lds(brow, hh), oacc[t2]);
        }
        WAIT_DS0();                 // pa consumed before next pbuf store
        __syncthreads();            // all waves done with buf before overwrite
    }

    // ---- normalize, store Hctx [B,S,D] bf16
#pragma unroll
    for (int t2 = 0; t2 < 4; ++t2) {
#pragma unroll
        for (int r = 0; r < 8; ++r) {
            int s = q0 + hh * 8 + r;
            int e = head * DHsz + t2 * 16 + lm;
            float v = oacc[t2][r] / lrow[r];
            Hctx[((size_t)b * Ssz + s) * Dsz + e] = (__bf16)v;
        }
    }
}

// ---------------------------------------------------------------------------
// LayerNorm over D=1024 per row
// ---------------------------------------------------------------------------
__global__ void k_layernorm(const float* __restrict__ resid,
                            const float* __restrict__ w,
                            const float* __restrict__ bias,
                            float* __restrict__ out) {
    __shared__ float s_sum[256];
    __shared__ float s_sq[256];
    const int tid = threadIdx.x;
    const float* rp = resid + (size_t)blockIdx.x * Dsz;

    float s = 0.f, s2 = 0.f;
#pragma unroll
    for (int i = tid; i < Dsz; i += 256) {
        float v = rp[i];
        s += v; s2 += v * v;
    }
    s_sum[tid] = s; s_sq[tid] = s2;
    __syncthreads();
    for (int o = 128; o > 0; o >>= 1) {
        if (tid < o) { s_sum[tid] += s_sum[tid + o]; s_sq[tid] += s_sq[tid + o]; }
        __syncthreads();
    }
    float mu  = s_sum[0] * (1.f / Dsz);
    float var = s_sq[0] * (1.f / Dsz) - mu * mu;
    float rstd = rsqrtf(var + LN_EPS);

    float* op = out + (size_t)blockIdx.x * Dsz;
#pragma unroll
    for (int i = tid; i < Dsz; i += 256) {
        op[i] = (rp[i] - mu) * rstd * w[i] + bias[i];
    }
}

// ---------------------------------------------------------------------------
// Host-side launcher
// ---------------------------------------------------------------------------
extern "C" void kernel_launch(void* const* d_in, const int* in_sizes, int n_in,
                              void* d_out, int out_size, void* d_ws, size_t ws_size,
                              hipStream_t stream) {
    const float* x    = (const float*)d_in[0];
    const int*   msk  = (const int*)  d_in[1];
    const float* Wq   = (const float*)d_in[2];
    const float* bq   = (const float*)d_in[3];
    const float* Wk   = (const float*)d_in[4];
    const float* bk   = (const float*)d_in[5];
    const float* Wv   = (const float*)d_in[6];
    const float* bv   = (const float*)d_in[7];
    const float* Wo   = (const float*)d_in[8];
    const float* bo   = (const float*)d_in[9];
    const float* ln_w = (const float*)d_in[10];
    const float* ln_b = (const float*)d_in[11];

    constexpr size_t MB = 1024ull * 1024ull;
    char* ws = (char*)d_ws;
    __bf16* xbf = (__bf16*)(ws + 0);        // 16 MB [M,D]
    __bf16* wqb = (__bf16*)(ws + 16 * MB);
    __bf16* wkb = (__bf16*)(ws + 18 * MB);
    __bf16* wvb = (__bf16*)(ws + 20 * MB);
    __bf16* wob = (__bf16*)(ws + 22 * MB);
    __bf16* Qb  = (__bf16*)(ws + 24 * MB);  // 16 MB [B,H,S,DH]
    __bf16* Kb  = (__bf16*)(ws + 40 * MB);  // 16 MB [B,H,S,DH]
    __bf16* Vtb = (__bf16*)(ws + 56 * MB);  // 16 MB [B,H,DH,S]
    __bf16* Hcb = (__bf16*)(ws + 72 * MB);  // 16 MB [M,D]
    float*  res = (float*) (ws + 24 * MB);  // 32 MB, aliases dead Q+K

    k_f32_to_bf16<<<2048, 256, 0, stream>>>(x,  xbf, (long)Msz * Dsz);
    k_f32_to_bf16<<<1024, 256, 0, stream>>>(Wq, wqb, (long)Dsz * Dsz);
    k_f32_to_bf16<<<1024, 256, 0, stream>>>(Wk, wkb, (long)Dsz * Dsz);
    k_f32_to_bf16<<<1024, 256, 0, stream>>>(Wv, wvb, (long)Dsz * Dsz);
    k_f32_to_bf16<<<1024, 256, 0, stream>>>(Wo, wob, (long)Dsz * Dsz);

    dim3 ggrid(Msz / 128, Dsz / 64);
    k_gemm_qkv<<<ggrid, 256, 0, stream>>>(xbf, wqb, bq, Qb, 0);
    k_gemm_qkv<<<ggrid, 256, 0, stream>>>(xbf, wkb, bk, Kb, 0);
    k_gemm_qkv<<<ggrid, 256, 0, stream>>>(xbf, wvb, bv, Vtb, 1);

    k_flash_attn<<<Bsz * Hsz * (Ssz / 128), 256, 0, stream>>>(Qb, Kb, Vtb, msk, Hcb);

    k_gemm_oproj<<<ggrid, 256, 0, stream>>>(Hcb, wob, bo, x, res);

    k_layernorm<<<Msz, 256, 0, stream>>>(res, ln_w, ln_b, (float*)d_out);
}
